// GQA_74560632259157
// MI455X (gfx1250) — matrix operand order repaired and checked by reading, not compiled, not run on hardware
//
#include <hip/hip_runtime.h>

#define BD   2
#define SEQ  2048
#define DM   1024
#define HD   128
#define QH   16
#define KVH  4
#define GRP  4
#define MTOT (BD * SEQ)

typedef __attribute__((ext_vector_type(16))) __bf16       v16bf;
typedef __attribute__((ext_vector_type(8)))  float        v8f;
typedef __attribute__((ext_vector_type(4)))  float        v4f;
typedef __attribute__((ext_vector_type(4)))  unsigned int v4u;

union Frag {
    v16bf v;
    unsigned short u[16];
    v4u q[2];
};

__device__ __forceinline__ unsigned short f2bf(float f) {
    __bf16 h = (__bf16)f;                        // native cvt (RNE) on gfx1250
    return __builtin_bit_cast(unsigned short, h);
}

__device__ __forceinline__ float fast_exp2(float x) {
    return __builtin_amdgcn_exp2f(x);
}

// max-reduce/broadcast across each 16-lane half:
// xor1/xor2 via DPP quad_perm, xor4 via DPP8, xor8 via bpermute.
__device__ __forceinline__ float half_allmax(float x) {
    constexpr int DPP8_XOR4 =
        (4 << 0) | (5 << 3) | (6 << 6) | (7 << 9) |
        (0 << 12) | (1 << 15) | (2 << 18) | (3 << 21);
    int xi = __builtin_bit_cast(int, x);
    int t  = __builtin_amdgcn_update_dpp(xi, xi, 0xB1, 0xF, 0xF, true); // [1,0,3,2]
    x = fmaxf(x, __builtin_bit_cast(float, t));
    xi = __builtin_bit_cast(int, x);
    t  = __builtin_amdgcn_update_dpp(xi, xi, 0x4E, 0xF, 0xF, true);     // [2,3,0,1]
    x = fmaxf(x, __builtin_bit_cast(float, t));
    xi = __builtin_bit_cast(int, x);
    t  = __builtin_amdgcn_mov_dpp8(xi, DPP8_XOR4);
    x = fmaxf(x, __builtin_bit_cast(float, t));
    x = fmaxf(x, __shfl_xor(x, 8, 32));
    return x;
}

// ---------------------------------------------------------------------------
// Projection GEMM: out_bf16 = bf16( X[M x K] @ W[K x N] + bias[N] )
// Block tile 64x128, 8 waves (2x4), wave tile 32x32, K-step 32.
// STORET==false: out[row*N + col]; STORET==true: out[col*MTOT + row] (for V)
// ---------------------------------------------------------------------------
template <int N, bool STORET>
__global__ __launch_bounds__(256) void qkv_gemm_bf16(
    const float* __restrict__ X, const float* __restrict__ W,
    const float* __restrict__ bias, unsigned short* __restrict__ out)
{
    __shared__ unsigned short As[64][40];     // 64x32 A tile (+pad, rows 80B)
    __shared__ unsigned short BsT[128][40];   // B tile transposed: [col][k]

    const int tid  = threadIdx.x;
    const int lane = tid & 31;
    const int w    = tid >> 5;
    const int wm   = w >> 2;       // 0..1
    const int wn   = w & 3;        // 0..3
    const int lo   = lane & 15;
    const int hi   = lane >> 4;

    const int gm = blockIdx.y * 64;
    const int gn = blockIdx.x * 128;

    const v8f vzero = {0.f, 0.f, 0.f, 0.f, 0.f, 0.f, 0.f, 0.f};
    v8f acc[2][2];
    acc[0][0] = vzero; acc[0][1] = vzero; acc[1][0] = vzero; acc[1][1] = vzero;

    const int ar = tid >> 2, ac = (tid & 3) * 8;      // A staging: 8 floats
    const int br = tid >> 3, bc = (tid & 7) * 16;     // B staging: 16 floats

    for (int k0 = 0; k0 < DM; k0 += 32) {
        {
            const float* src = X + (unsigned)(gm + ar) * DM + k0 + ac;
            const v4f f0 = *(const v4f*)(src);
            const v4f f1 = *(const v4f*)(src + 4);
            Frag t;
#pragma unroll
            for (int i = 0; i < 4; ++i) {
                t.u[i]     = f2bf(f0[i]);
                t.u[4 + i] = f2bf(f1[i]);
            }
            *(v4u*)&As[ar][ac] = t.q[0];
        }
        {
            const float* src = W + (unsigned)(k0 + br) * N + gn + bc;
            v4f f[4];
#pragma unroll
            for (int c4 = 0; c4 < 4; ++c4) f[c4] = *(const v4f*)(src + c4 * 4);
#pragma unroll
            for (int c4 = 0; c4 < 4; ++c4)
#pragma unroll
                for (int i = 0; i < 4; ++i)
                    BsT[bc + c4 * 4 + i][br] = f2bf(f[c4][i]);
        }
        __syncthreads();

        Frag a[2], bfr[2];
#pragma unroll
        for (int mt = 0; mt < 2; ++mt) {
            const int row = wm * 32 + mt * 16 + lo;   // A layout: lane = M
            a[mt].q[0] = *(const v4u*)&As[row][hi * 8];
            a[mt].q[1] = *(const v4u*)&As[row][16 + hi * 8];
        }
#pragma unroll
        for (int nt = 0; nt < 2; ++nt) {
            const int col = wn * 32 + nt * 16 + lo;   // B layout: lane = N
            bfr[nt].q[0] = *(const v4u*)&BsT[col][hi * 8];
            bfr[nt].q[1] = *(const v4u*)&BsT[col][16 + hi * 8];
        }
#pragma unroll
        for (int mt = 0; mt < 2; ++mt)
#pragma unroll
            for (int nt = 0; nt < 2; ++nt)
                acc[mt][nt] = __builtin_amdgcn_wmma_f32_16x16x32_bf16(
                    false, a[mt].v, false, bfr[nt].v, (short)0, acc[mt][nt],
                    false, false);
        __syncthreads();
    }

#pragma unroll
    for (int mt = 0; mt < 2; ++mt)
#pragma unroll
        for (int nt = 0; nt < 2; ++nt) {
            const int row0 = gm + wm * 32 + mt * 16 + hi * 8;
            const int col  = gn + wn * 32 + nt * 16 + lo;
            const float bv = bias[col];
#pragma unroll
            for (int r = 0; r < 8; ++r) {
                const unsigned short ov = f2bf(acc[mt][nt][r] + bv);
                if (STORET) out[(unsigned)col * MTOT + (row0 + r)] = ov;
                else        out[(unsigned)(row0 + r) * N + col]    = ov;
            }
        }
}

// ---------------------------------------------------------------------------
// Flash attention per (batch, q-head): each wave owns a 16-query tile,
// iterates keys 32 at a time. S: 8 wmma, P@V: 8 wmma, row-sum: 1 wmma.
// V workspace is pre-transposed: vbT[channel][b*SEQ + t].
// ---------------------------------------------------------------------------
__global__ __launch_bounds__(256) void gqa_attn(
    const unsigned short* __restrict__ qb,
    const unsigned short* __restrict__ kb,
    const unsigned short* __restrict__ vbT,
    float* __restrict__ out)
{
    __shared__ unsigned short Pbuf[8][16 * 32];  // per-wave P re-layout buffer

    const int tid  = threadIdx.x;
    const int lane = tid & 31;
    const int w    = tid >> 5;
    const int lo   = lane & 15;
    const int hi   = lane >> 4;

    const int bi = blockIdx.z;
    const int h  = blockIdx.y;          // q head = g*KVH + kv
    const int g  = h >> 2;
    const int kv = h & 3;
    const int qbase = (blockIdx.x * 8 + w) * 16;

    const unsigned short* qrow = qb + (unsigned)(bi * SEQ + qbase) * (QH * HD) + h * HD;
    const unsigned short* khp  = kb + (unsigned)bi * SEQ * (KVH * HD) + kv * HD;
    const unsigned short* vhp  = vbT + (unsigned)(kv * HD) * MTOT + bi * SEQ;

    // Q fragments: 4 chunks of K=32 over head_dim 128, kept in VGPRs
    Frag qf[4];
#pragma unroll
    for (int c = 0; c < 4; ++c) {
        const unsigned short* p = qrow + (unsigned)lo * (QH * HD) + c * 32 + hi * 8;
        qf[c].q[0] = *(const v4u*)(p);
        qf[c].q[1] = *(const v4u*)(p + 16);
    }

    // all-ones bf16 B fragment for WMMA row-sums
    Frag ones;
#pragma unroll
    for (int i = 0; i < 16; ++i) ones.u[i] = 0x3F80;

    const v8f vzero = {0.f, 0.f, 0.f, 0.f, 0.f, 0.f, 0.f, 0.f};
    v8f o[8];
#pragma unroll
    for (int jn = 0; jn < 8; ++jn) o[jn] = vzero;
    v8f lsum = vzero;                 // row-sums accumulated via WMMA
    float m_i[8];
#pragma unroll
    for (int r = 0; r < 8; ++r) m_i[r] = -__builtin_inff();

    // 1/sqrt(128) * log2(e): softmax in exp2 domain
    const float kscale = 0.08838834764831845f * 1.4426950408889634f;

    for (int t0 = 0; t0 < SEQ; t0 += 32) {
        // ---- S = Q @ K^T for 32 keys (two 16x16 tiles) ----
        v8f s0 = vzero, s1 = vzero;
#pragma unroll
        for (int c = 0; c < 4; ++c) {
            Frag k0f, k1f;
            const unsigned short* kp0 = khp + (unsigned)(t0 + lo)      * (KVH * HD) + c * 32 + hi * 8;
            const unsigned short* kp1 = khp + (unsigned)(t0 + 16 + lo) * (KVH * HD) + c * 32 + hi * 8;
            k0f.q[0] = *(const v4u*)(kp0);
            k0f.q[1] = *(const v4u*)(kp0 + 16);
            k1f.q[0] = *(const v4u*)(kp1);
            k1f.q[1] = *(const v4u*)(kp1 + 16);
            s0 = __builtin_amdgcn_wmma_f32_16x16x32_bf16(false, qf[c].v, false, k0f.v,
                                                         (short)0, s0, false, false);
            s1 = __builtin_amdgcn_wmma_f32_16x16x32_bf16(false, qf[c].v, false, k1f.v,
                                                         (short)0, s1, false, false);
        }

        // ---- online softmax: only the row-max needs cross-lane reduce ----
        float p0v[8], p1v[8];
        v8f corrv;
#pragma unroll
        for (int r = 0; r < 8; ++r) {
            const float a0 = s0[r] * kscale, a1 = s1[r] * kscale;
            const float mx = half_allmax(fmaxf(a0, a1));
            const float nm = fmaxf(m_i[r], mx);
            corrv[r] = fast_exp2(m_i[r] - nm);
            p0v[r]   = fast_exp2(a0 - nm);
            p1v[r]   = fast_exp2(a1 - nm);
            m_i[r]   = nm;
        }
        lsum = lsum * corrv;
#pragma unroll
        for (int jn = 0; jn < 8; ++jn) o[jn] = o[jn] * corrv;  // v_pk_mul_f32

        // ---- re-layout P: C-layout -> A-fragment layout via per-wave LDS ----
#pragma unroll
        for (int r = 0; r < 8; ++r) {
            Pbuf[w][(hi * 8 + r) * 32 + lo]      = f2bf(p0v[r]);
            Pbuf[w][(hi * 8 + r) * 32 + 16 + lo] = f2bf(p1v[r]);
        }
        asm volatile("s_wait_dscnt 0" ::: "memory");
        Frag pf;
        pf.q[0] = *(const v4u*)&Pbuf[w][lo * 32 + hi * 8];
        pf.q[1] = *(const v4u*)&Pbuf[w][lo * 32 + 16 + hi * 8];

        // ---- row-sums of P via the matrix pipe: lsum += P @ ones ----
        lsum = __builtin_amdgcn_wmma_f32_16x16x32_bf16(false, pf.v, false, ones.v,
                                                       (short)0, lsum, false, false);

        // ---- O += P @ V (8 n-tiles over head_dim), V pre-transposed ----
#pragma unroll
        for (int jn = 0; jn < 8; ++jn) {
            Frag vf;
            const unsigned short* vp = vhp + (unsigned)(jn * 16 + lo) * MTOT + t0 + hi * 8;
            vf.q[0] = *(const v4u*)(vp);
            vf.q[1] = *(const v4u*)(vp + 16);
            o[jn] = __builtin_amdgcn_wmma_f32_16x16x32_bf16(false, pf.v, false, vf.v,
                                                            (short)0, o[jn], false, false);
        }
    }

    // epilogue: normalize (vector rcp*mul), write fp32, channel order (kv,g,hd)
    v8f rinvv;
#pragma unroll
    for (int r = 0; r < 8; ++r) rinvv[r] = 1.0f / lsum[r];
    const int outoff = (kv * GRP + g) * HD;
#pragma unroll
    for (int jn = 0; jn < 8; ++jn) {
        const v8f ov = o[jn] * rinvv;
#pragma unroll
        for (int r = 0; r < 8; ++r) {
            const int row = qbase + hi * 8 + r;
            const int col = outoff + jn * 16 + lo;
            out[(unsigned)(bi * SEQ + row) * (QH * HD) + col] = ov[r];
        }
    }
}

// ---------------------------------------------------------------------------
extern "C" void kernel_launch(void* const* d_in, const int* in_sizes, int n_in,
                              void* d_out, int out_size, void* d_ws, size_t ws_size,
                              hipStream_t stream) {
    const float* x  = (const float*)d_in[0];
    const float* Wq = (const float*)d_in[1];
    const float* bq = (const float*)d_in[2];
    const float* Wk = (const float*)d_in[3];
    const float* bk = (const float*)d_in[4];
    const float* Wv = (const float*)d_in[5];
    const float* bv = (const float*)d_in[6];
    float* out = (float*)d_out;

    // workspace: bf16 Q (16MB), K (4MB), V-transposed (4MB)
    unsigned short* qb  = (unsigned short*)d_ws;
    unsigned short* kb  = qb + (size_t)MTOT * QH * HD;
    unsigned short* vbt = kb + (size_t)MTOT * KVH * HD;

    dim3 blk(256);
    qkv_gemm_bf16<QH * HD, false>
        <<<dim3((QH * HD) / 128, MTOT / 64), blk, 0, stream>>>(x, Wq, bq, qb);
    qkv_gemm_bf16<KVH * HD, false>
        <<<dim3((KVH * HD) / 128, MTOT / 64), blk, 0, stream>>>(x, Wk, bk, kb);
    qkv_gemm_bf16<KVH * HD, true>
        <<<dim3((KVH * HD) / 128, MTOT / 64), blk, 0, stream>>>(x, Wv, bv, vbt);
    gqa_attn<<<dim3(SEQ / 128, QH, BD), blk, 0, stream>>>(qb, kb, vbt, out);
}
